// JumpEulerForwardCuda_79834852098115
// MI455X (gfx1250) — matrix operand order repaired and checked by reading, not compiled, not run on hardware
//
#include <hip/hip_runtime.h>

typedef __attribute__((ext_vector_type(16))) _Float16 v16h;
typedef __attribute__((ext_vector_type(8)))  float    v8f;

#define NSIM    16384
#define NSTEPS  500
#define DDIM    10
#define HDIM    64
#define DTC     0.03f
#define SQRTDT  0.17320508075688772f

#define WAVES_PER_BLOCK 4
#define TPB   (WAVES_PER_BLOCK * 32)
#define NFRAG 28                      // 14 B-fragments per MLP x 2 MLPs
#define FRAG_HALVES (NFRAG * 32 * 16) // per-lane 16 halves (32B) per fragment

__device__ __forceinline__ void lds_fence() {
  asm volatile("s_wait_dscnt 0x0" ::: "memory");
  __builtin_amdgcn_wave_barrier();
}

__device__ __forceinline__ v8f wmma_f16(v16h a, v16h b, v8f c) {
  // (neg_a, A, neg_b, B, c_mod, C, reuse_a, reuse_b)
  return __builtin_amdgcn_wmma_f32_16x16x32_f16(false, a, false, b, (short)0, c,
                                                false, false);
}

// Async global->LDS (GVS mode): vdst = LDS byte offset, vaddr = 32-bit byte
// offset, saddr = 64-bit base. Tracked with ASYNCcnt.
__device__ __forceinline__ void async_ld_b32(void* lds_dst, const float* base,
                                             uint32_t byte_off) {
  uint32_t lds_off = (uint32_t)(uintptr_t)lds_dst; // low 32 bits = LDS offset
  asm volatile("global_load_async_to_lds_b32 %0, %1, %2"
               :: "v"(lds_off), "v"(byte_off), "s"(base) : "memory");
}

// ---------------------------------------------------------------------------
// Setup: swizzle the six weight matrices into WMMA B-fragment layout in d_ws.
// B 32x16 f16 layout: lane L holds column N = L%16; element i holds
// K = i + 8*(L/16) for i<8, K = (i-8)+16+8*(L/16) for i>=8 within the 32-wide K chunk.
// ---------------------------------------------------------------------------
__global__ void frag_setup(const float* __restrict__ dW0, const float* __restrict__ dW1,
                           const float* __restrict__ dW2, const float* __restrict__ jW0,
                           const float* __restrict__ jW1, const float* __restrict__ jW2,
                           _Float16* __restrict__ frags)
{
  const int f    = blockIdx.x;   // 0..27
  const int lane = threadIdx.x;  // 0..31
  const int mlp   = f / 14;
  const int local = f % 14;

  const float* W0 = mlp ? jW0 : dW0;
  const float* W1 = mlp ? jW1 : dW1;
  const float* W2 = mlp ? jW2 : dW2;

  const float* W; int inDim, outDim, ntile, kc;
  if (local < 4)       { W = W0; inDim = DDIM; outDim = HDIM; ntile = local;      kc = 0; }
  else if (local < 12) { W = W1; inDim = HDIM; outDim = HDIM; int t = local - 4;
                         kc = t >> 2; ntile = t & 3; }
  else                 { W = W2; inDim = HDIM; outDim = DDIM; ntile = 0; kc = local - 12; }

  const int n    = ntile * 16 + (lane & 15);
  const int half = lane >> 4;
  _Float16* dst  = frags + ((size_t)f * 32 + lane) * 16;

  for (int i = 0; i < 16; ++i) {
    int kg = kc * 32 + ((i < 8) ? (i + 8 * half) : (i + 8 + 8 * half));
    float v = (kg < inDim && n < outDim) ? W[kg * outDim + n] : 0.0f;
    dst[i] = (_Float16)v;
  }
}

// ---------------------------------------------------------------------------
// Main persistent kernel: one wave per 16-sim tile, all 500 steps on-chip.
// Noise is double-buffered and prefetched with async-to-LDS loads.
// ---------------------------------------------------------------------------
struct WaveBuf {
  float    Zt[256];     // 16x16 f32 state tile (cols >= 10 stay zero)
  _Float16 Ha[1024];    // 16x64 activations (layer-0 out)
  _Float16 Hb[1024];    // 16x64 activations (layer-1 out)
  float    ND[2][160];  // noise_d tile   16x10, double-buffered
  float    NBD[2][32];  // noise_bd tile  16x2
  float    P[2][16];    // poisson counts 16x1
};

struct Smem {
  alignas(32) _Float16 frags[FRAG_HALVES]; // 28 KiB of pre-swizzled B fragments
  float bias0[2][64];
  float bias1[2][64];
  float bias2[2][16];
  float mean[16];
  float cov[112];   // 10x10 used, row-major covHalf[k*10+n]
  float dif[32];    // 2x10 used,  row-major diffusion[b*10+n]
  WaveBuf wb[WAVES_PER_BLOCK];
};

// A 16x32 f16 fragment gather: lane L -> row L%16,
// element i -> K = i + 8*(L/16) for i<8, K = (i-8)+16+8*(L/16) for i>=8.
__device__ __forceinline__ v16h gatherA(const _Float16* __restrict__ H, int kc,
                                        int nl, int half)
{
  union { v16h v; _Float16 e[16]; } u;
#pragma unroll
  for (int i = 0; i < 16; ++i) {
    int k = (i < 8) ? (i + 8 * half) : (i + 8 + 8 * half);
    u.e[i] = H[nl * HDIM + kc * 32 + k];
  }
  return u.v;
}

// Issue exactly 7 async-load instructions (uniform per wave, EXEC all-1s).
__device__ __forceinline__ void issue_noise(WaveBuf& wb, int buf, int lane, size_t nb,
                                            const float* __restrict__ nd,
                                            const float* __restrict__ nbd,
                                            const float* __restrict__ pois)
{
#pragma unroll
  for (int t = 0; t < 5; ++t) {
    int j = lane + 32 * t;
    async_ld_b32(&wb.ND[buf][j], nd, (uint32_t)((nb * DDIM + j) * 4));
  }
  async_ld_b32(&wb.NBD[buf][lane], nbd, (uint32_t)((nb * 2 + lane) * 4));
  int pl = lane & 15; // lanes 16..31 duplicate lanes 0..15 (same data, same addr)
  async_ld_b32(&wb.P[buf][pl], pois, (uint32_t)((nb + pl) * 4));
}

__global__ __launch_bounds__(TPB) void sde_jump_euler(
    const float* __restrict__ z0,
    const float* __restrict__ db0, const float* __restrict__ db1, const float* __restrict__ db2,
    const float* __restrict__ jb0, const float* __restrict__ jb1, const float* __restrict__ jb2,
    const float* __restrict__ meanv, const float* __restrict__ covHalf,
    const float* __restrict__ dif,
    const float* __restrict__ pois, const float* __restrict__ nbd,
    const float* __restrict__ nd,
    const _Float16* __restrict__ fragsG, float* __restrict__ out)
{
  __shared__ Smem sm;
  const int tid = threadIdx.x;

  // Stage fragments + parameters into LDS once.
  {
    const uint32_t* s = (const uint32_t*)fragsG;
    uint32_t* d = (uint32_t*)sm.frags;
    for (int i = tid; i < FRAG_HALVES / 2; i += TPB) d[i] = s[i];
  }
  for (int i = tid; i < 64; i += TPB) {
    sm.bias0[0][i] = db0[i]; sm.bias0[1][i] = jb0[i];
    sm.bias1[0][i] = db1[i]; sm.bias1[1][i] = jb1[i];
  }
  for (int i = tid; i < 16; i += TPB) {
    sm.bias2[0][i] = (i < DDIM) ? db2[i]   : 0.0f;
    sm.bias2[1][i] = (i < DDIM) ? jb2[i]   : 0.0f;
    sm.mean[i]     = (i < DDIM) ? meanv[i] : 0.0f;
  }
  for (int i = tid; i < 100; i += TPB) sm.cov[i] = covHalf[i];
  for (int i = tid; i < 20;  i += TPB) sm.dif[i] = dif[i];
  __syncthreads();

  const int wave = tid >> 5, lane = tid & 31;
  const int half = lane >> 4, nl = lane & 15;
  WaveBuf& wb = sm.wb[wave];
  const int simBase = (blockIdx.x * WAVES_PER_BLOCK + wave) * 16;

  // Prefetch step-0 noise into buffer 0 (7 async ops).
  issue_noise(wb, 0, lane, (size_t)simBase, nd, nbd, pois);

  // Initialize state tile and write path[:, 0, :].
  for (int i = lane; i < 256; i += 32) wb.Zt[i] = 0.0f;
  for (int j = lane; j < 160; j += 32) {
    int m = j / DDIM, n = j % DDIM;
    float z = z0[(simBase + m) * DDIM + n];
    wb.Zt[m * 16 + n] = z;
    __builtin_nontemporal_store(z, &out[((size_t)(simBase + m) * (NSTEPS + 1)) * DDIM + n]);
  }
  lds_fence();

  for (int step = 0; step < NSTEPS; ++step) {
    const int cur = step & 1;

    // Software pipeline: issue next step's noise, then wait for current batch.
    // Async loads retire in order, so asynccnt <= 7 (the batch just issued)
    // implies the previous batch has fully landed in LDS.
    if (step + 1 < NSTEPS) {
      issue_noise(wb, cur ^ 1, lane, (size_t)(step + 1) * NSIM + simBase, nd, nbd, pois);
      asm volatile("s_wait_asynccnt 0x7" ::: "memory");
    } else {
      asm volatile("s_wait_asynccnt 0x0" ::: "memory");
    }
    __builtin_amdgcn_wave_barrier();

    // State A fragment: f32 -> f16, K padded 10 -> 32.
    v16h Az;
    {
      union { v16h v; _Float16 e[16]; } u;
#pragma unroll
      for (int i = 0; i < 16; ++i) {
        int k = (i < 8) ? (i + 8 * half) : (i + 8 + 8 * half);
        u.e[i] = (k < 16) ? (_Float16)wb.Zt[nl * 16 + k] : (_Float16)0.0f;
      }
      Az = u.v;
    }

    float mout[2][8];
#pragma unroll 1
    for (int mlp = 0; mlp < 2; ++mlp) {
      const int fb = mlp * 14;

      // Layer 0: z(16x10) @ W0(10x64), bias + LeakyReLU -> Ha (f16)
#pragma unroll
      for (int nt = 0; nt < 4; ++nt) {
        v8f acc = {};
        v16h b = *(const v16h*)(sm.frags + ((fb + nt) * 32 + lane) * 16);
        acc = wmma_f16(Az, b, acc);
        const int n  = nt * 16 + nl;
        const float bb = sm.bias0[mlp][n];
#pragma unroll
        for (int r = 0; r < 8; ++r) {
          float v = acc[r] + bb;
          v = (v > 0.0f) ? v : 0.01f * v;
          wb.Ha[(r + 8 * half) * HDIM + n] = (_Float16)v;
        }
      }
      lds_fence();

      // Layer 1: Ha(16x64) @ W1(64x64), two K chunks, -> Hb
      v16h a0 = gatherA(wb.Ha, 0, nl, half);
      v16h a1 = gatherA(wb.Ha, 1, nl, half);
#pragma unroll
      for (int nt = 0; nt < 4; ++nt) {
        v8f acc = {};
        v16h b0 = *(const v16h*)(sm.frags + ((fb + 4 + nt) * 32 + lane) * 16);
        v16h b1 = *(const v16h*)(sm.frags + ((fb + 8 + nt) * 32 + lane) * 16);
        acc = wmma_f16(a0, b0, acc);
        acc = wmma_f16(a1, b1, acc);
        const int n  = nt * 16 + nl;
        const float bb = sm.bias1[mlp][n];
#pragma unroll
        for (int r = 0; r < 8; ++r) {
          float v = acc[r] + bb;
          v = (v > 0.0f) ? v : 0.01f * v;
          wb.Hb[(r + 8 * half) * HDIM + n] = (_Float16)v;
        }
      }
      lds_fence();

      // Layer 2: Hb(16x64) @ W2(64x10), linear out (N padded to 16)
      v16h c0 = gatherA(wb.Hb, 0, nl, half);
      v16h c1 = gatherA(wb.Hb, 1, nl, half);
      v8f acc = {};
      v16h b0 = *(const v16h*)(sm.frags + ((fb + 12) * 32 + lane) * 16);
      v16h b1 = *(const v16h*)(sm.frags + ((fb + 13) * 32 + lane) * 16);
      acc = wmma_f16(c0, b0, acc);
      acc = wmma_f16(c1, b1, acc);
      const float b2v = sm.bias2[mlp][nl];
#pragma unroll
      for (int r = 0; r < 8; ++r) mout[mlp][r] = acc[r] + b2v;
    }

    // Elementwise jump-diffusion update in the C/D layout:
    // VGPR r -> row m = r + 8*half, column n = lane % 16 (valid for n < 10).
    if (nl < DDIM) {
#pragma unroll
      for (int r = 0; r < 8; ++r) {
        const int m = r + 8 * half;
        float z  = wb.Zt[m * 16 + nl];
        float p  = wb.P[cur][m];
        float sq = sqrtf(p);
        float cv = 0.0f;
#pragma unroll
        for (int k = 0; k < DDIM; ++k) cv += wb.ND[cur][m * DDIM + k] * sm.cov[k * DDIM + nl];
        float df   = wb.NBD[cur][m * 2] * sm.dif[nl] + wb.NBD[cur][m * 2 + 1] * sm.dif[DDIM + nl];
        float jamp = p * sm.mean[nl] + sq * cv;
        float zn   = z + mout[0][r] * DTC + SQRTDT * df + jamp * mout[1][r];
        __builtin_nontemporal_store(
            zn, &out[((size_t)(simBase + m) * (NSTEPS + 1) + step + 1) * DDIM + nl]);
        wb.Zt[m * 16 + nl] = zn;
      }
    }
    lds_fence();
  }
}

extern "C" void kernel_launch(void* const* d_in, const int* in_sizes, int n_in,
                              void* d_out, int out_size, void* d_ws, size_t ws_size,
                              hipStream_t stream) {
  (void)in_sizes; (void)n_in; (void)out_size; (void)ws_size;

  const float* z0   = (const float*)d_in[0];
  const float* dW0  = (const float*)d_in[1];
  const float* db0  = (const float*)d_in[2];
  const float* dW1  = (const float*)d_in[3];
  const float* db1  = (const float*)d_in[4];
  const float* dW2  = (const float*)d_in[5];
  const float* db2  = (const float*)d_in[6];
  const float* jW0  = (const float*)d_in[7];
  const float* jb0  = (const float*)d_in[8];
  const float* jW1  = (const float*)d_in[9];
  const float* jb1  = (const float*)d_in[10];
  const float* jW2  = (const float*)d_in[11];
  const float* jb2  = (const float*)d_in[12];
  const float* mean = (const float*)d_in[13];
  const float* cov  = (const float*)d_in[14];
  const float* dif  = (const float*)d_in[15];
  const float* pois = (const float*)d_in[16];
  const float* nbd  = (const float*)d_in[17];
  const float* nd   = (const float*)d_in[18];

  _Float16* frags = (_Float16*)d_ws;
  float* out = (float*)d_out;

  frag_setup<<<NFRAG, 32, 0, stream>>>(dW0, dW1, dW2, jW0, jW1, jW2, frags);

  const int blocks = NSIM / (WAVES_PER_BLOCK * 16); // 256
  sde_jump_euler<<<blocks, TPB, 0, stream>>>(
      z0, db0, db1, db2, jb0, jb1, jb2, mean, cov, dif,
      pois, nbd, nd, frags, out);
}
